// FFNerfModel_3092376453816
// MI455X (gfx1250) — compile-verified, matrix operand order checked by reading, not run on hardware
//
#include <hip/hip_runtime.h>
#include <hip/hip_bf16.h>

typedef __attribute__((ext_vector_type(16))) _Float16 v16h;
typedef __attribute__((ext_vector_type(8)))  _Float16 v8h;
typedef __attribute__((ext_vector_type(8)))  float    v8f;

#define ROWS    128     // points per workgroup (8 waves x 16 rows)
#define THREADS 256
#define ACT_W   200     // 192 cols + 8-half skew (row stride 400B = 100 dwords -> conflict-free A loads)
#define WBUF_HALVES 24576   // max padded layer: 192 x 128 (one weight buffer)

// ---- d_ws layout: pre-swizzled f16 weight blocks (halves) ----
#define SZ_64x128   (64 * 128)
#define SZ_128x128  (128 * 128)
#define SZ_192x128  (192 * 128)
#define SZ_128x144  (128 * 144)
#define SZ_160x64   (160 * 64)
#define OFF_W10 0
#define OFF_W11 (OFF_W10 + SZ_64x128)
#define OFF_W12 (OFF_W11 + SZ_128x128)
#define OFF_W13 (OFF_W12 + SZ_128x128)
#define OFF_W14 (OFF_W13 + SZ_128x128)
#define OFF_W20 (OFF_W14 + SZ_128x128)
#define OFF_W21 (OFF_W20 + SZ_192x128)
#define OFF_W22 (OFF_W21 + SZ_128x128)
#define OFF_W23 (OFF_W22 + SZ_128x128)
#define OFF_W30 (OFF_W23 + SZ_128x144)
#define WS_TOTAL (OFF_W30 + SZ_160x64)      // 159744 halves = 312 KB

// ---------------------------------------------------------------------------
// One-time pre-pass: f32 row-major [Kreal][Nreal] -> f16 B-fragment layout in
// global scratch.  block(kt,nt) = 512 halves split into two 256-half chunk
// planes; lane ln = (n&15)|(hi<<4) owns 8 contiguous halves per plane
// (chunk starts stride 4 dwords -> LDS banks 4*ln mod 64 when later loaded).
// Register half H maps to k = kt*32 + (H&7) + 8*hi + 16*(H>=8)  (ISA 05_wmma).
// ---------------------------------------------------------------------------
template<int Kpad, int Npad>
__global__ __launch_bounds__(THREADS) void preswizzle(const float* __restrict__ W,
                                                      _Float16* __restrict__ dst,
                                                      int Kreal, int Nreal) {
    constexpr int ntiles = Npad >> 4;
    const int idx = blockIdx.x * THREADS + threadIdx.x;   // grid covers Kpad*Npad exactly
    const int k = idx / Npad;
    const int n = idx - k * Npad;
    float v = 0.0f;
    if (k < Kreal && n < Nreal) v = W[k * Nreal + n];
    const int kp  = k & 31;
    const int hi  = (kp >> 3) & 1;
    const int c   = (kp >> 4) & 1;
    const int h7  = kp & 7;
    const int blk = (k >> 5) * ntiles + (n >> 4);
    const int ln  = (n & 15) | (hi << 4);
    dst[(blk << 9) + (c << 8) + (ln << 3) + h7] = (_Float16)v;
}

// ---------------------------------------------------------------------------
// Async copy of a pre-swizzled weight block (global -> LDS), 16B per lane per
// op, tracked by ASYNCcnt.  Issue-and-forget: overlaps the previous layer's
// WMMA compute.  LDS byte offset = low 32 bits of the generic pointer.
// ---------------------------------------------------------------------------
template<int HALVES>
__device__ __forceinline__ void async_stage(_Float16* ldsDst, const _Float16* gsrc, int tid) {
    constexpr int CHUNKS = HALVES / (THREADS * 8);
    const unsigned lds = (unsigned)(unsigned long long)ldsDst + (unsigned)tid * 16u;
    unsigned long long g = (unsigned long long)(gsrc + tid * 8);
#pragma unroll
    for (int i = 0; i < CHUNKS; ++i) {
        asm volatile("global_load_async_to_lds_b128 %0, %1, off"
                     :: "v"(lds + (unsigned)(i * THREADS * 16)),
                        "v"(g + (unsigned long long)(i * THREADS * 16))
                     : "memory");
    }
}

__device__ __forceinline__ void pipe_barrier() {
    asm volatile("s_wait_asynccnt 0" ::: "memory");   // our async copies landed in LDS
    __syncthreads();                                  // everyone done reading old buffer
}

// A fragment (16x32 f16): lane (l&15)=row, hi=(l>=16); halves 0..7 from
// k=kbase+8*hi.., halves 8..15 from k=kbase+16+8*hi..  -> two 16B LDS loads.
__device__ __forceinline__ v16h ld_afrag(const _Float16* actBase, int row, int kbase, int hi) {
    const _Float16* p = actBase + row * ACT_W + kbase + (hi << 3);
    v8h lo = *(const v8h*)p;
    v8h hh = *(const v8h*)(p + 16);
    v16h r;
#pragma unroll
    for (int i = 0; i < 8; ++i) { r[i] = lo[i]; r[8 + i] = hh[i]; }
    return r;
}

// B fragment from the chunk-plane layout: two 16B loads, 256 halves apart.
__device__ __forceinline__ v16h ld_bfrag(const _Float16* wbuf, int blk, int lane) {
    const _Float16* p = wbuf + (blk << 9) + (lane << 3);
    v8h lo = *(const v8h*)p;
    v8h hh = *(const v8h*)(p + 256);
    v16h r;
#pragma unroll
    for (int i = 0; i < 8; ++i) { r[i] = lo[i]; r[8 + i] = hh[i]; }
    return r;
}

// relu in ONE VALU op: med3(v, 0, +MAX) clamps to [0, inf)
__device__ __forceinline__ float relu1(float v) {
    return __builtin_amdgcn_fmed3f(v, 0.0f, 3.402823466e38f);
}

// Convert+store a D column pair: v_cvt_pk_rtz_f16_f32 packs 2 rows in 1 op.
__device__ __forceinline__ void store_pair(_Float16* q, float v0, float v1) {
    auto pk = __builtin_amdgcn_cvt_pkrtz(v0, v1);   // __fp16 ext_vector(2)
    q[0]     = (_Float16)pk[0];
    q[ACT_W] = (_Float16)pk[1];
}

// One dense layer on a wave's 16-row tile.  kt-OUTER / nt-INNER with all NT
// accumulators live: consecutive v_wmma ops are independent (no D->C chain,
// no D->A/B RAW) so no hazard padding; B-frag loads pipeline ahead.  A frags
// fully register-resident before any store -> in-place update is safe.
template<int KT, int NT, bool RELU>
__device__ __forceinline__ void mlp_layer(const _Float16* actIn, _Float16* actOut,
                                          const _Float16* wbuf, int lane, int row16) {
    const int l15 = lane & 15;
    const int hi  = lane >> 4;
    const int row = row16 + l15;
    v16h a[KT];
#pragma unroll
    for (int kt = 0; kt < KT; ++kt) a[kt] = ld_afrag(actIn, row, kt * 32, hi);
    v8f acc[NT];
#pragma unroll
    for (int nt = 0; nt < NT; ++nt) acc[nt] = {};
#pragma unroll
    for (int kt = 0; kt < KT; ++kt) {
#pragma unroll
        for (int nt = 0; nt < NT; ++nt) {
            v16h b = ld_bfrag(wbuf, kt * NT + nt, lane);
            acc[nt] = __builtin_amdgcn_wmma_f32_16x16x32_f16(false, a[kt], false, b,
                                                             (short)0, acc[nt], false, false);
        }
    }
    // D layout: VGPR i, lanes 0-15 -> (M=i, N=l15); lanes 16-31 -> (M=i+8)
#pragma unroll
    for (int nt = 0; nt < NT; ++nt) {
#pragma unroll
        for (int i = 0; i < 8; i += 2) {
            float v0 = acc[nt][i], v1 = acc[nt][i + 1];
            if (RELU) { v0 = relu1(v0); v1 = relu1(v1); }
            store_pair(actOut + (row16 + i + (hi << 3)) * ACT_W + nt * 16 + l15, v0, v1);
        }
    }
}

__global__ __launch_bounds__(THREADS) void nerf_fused(
    const float* __restrict__ pos,  const float* __restrict__ dirs,
    const _Float16* __restrict__ ws, const float* __restrict__ w31,
    float* __restrict__ colorOut, float* __restrict__ sigmaOut, int nPts)
{
    extern __shared__ _Float16 smem[];
    _Float16* act  = smem;                  // ROWS * ACT_W; cols 0..127 hidden, 128..191 x_emb
    _Float16* demb = act  + ROWS * ACT_W;   // ROWS * 32 (d_emb stash)
    _Float16* wb0  = demb + ROWS * 32;      // WBUF_HALVES (ping)
    _Float16* wb1  = wb0  + WBUF_HALVES;    // WBUF_HALVES (pong)

    const int tid   = threadIdx.x;
    const int lane  = tid & 31;
    const int wave  = tid >> 5;
    const int row16 = wave << 4;
    const long long base = (long long)blockIdx.x * ROWS;

    // kick off layer-0 weights while we do posenc trig
    async_stage<SZ_64x128>(wb0, ws + OFF_W10, tid);

    // ---- positional encodings (f32 math, stored f16) ----
    if (tid < ROWS) {
        const int r = tid;
        long long p = base + r; if (p >= nPts) p = nPts - 1;
        const float x0 = pos[p * 3 + 0], x1 = pos[p * 3 + 1], x2 = pos[p * 3 + 2];
        _Float16* dst = act + r * ACT_W + 128;
        dst[0] = (_Float16)x0; dst[1] = (_Float16)x1; dst[2] = (_Float16)x2;
        float s = 1.0f;
#pragma unroll
        for (int j = 0; j < 10; ++j) {
            dst[3 + 6 * j + 0] = (_Float16)__sinf(s * x0);
            dst[3 + 6 * j + 1] = (_Float16)__sinf(s * x1);
            dst[3 + 6 * j + 2] = (_Float16)__sinf(s * x2);
            dst[3 + 6 * j + 3] = (_Float16)__cosf(s * x0);
            dst[3 + 6 * j + 4] = (_Float16)__cosf(s * x1);
            dst[3 + 6 * j + 5] = (_Float16)__cosf(s * x2);
            s *= 2.0f;
        }
        dst[63] = (_Float16)0.0f;                       // pad col 191
    } else {
        const int r = tid - ROWS;
        long long p = base + r; if (p >= nPts) p = nPts - 1;
        const float x0 = dirs[p * 3 + 0], x1 = dirs[p * 3 + 1], x2 = dirs[p * 3 + 2];
        _Float16* dst = demb + r * 32;
        dst[0] = (_Float16)x0; dst[1] = (_Float16)x1; dst[2] = (_Float16)x2;
        float s = 1.0f;
#pragma unroll
        for (int j = 0; j < 4; ++j) {
            dst[3 + 6 * j + 0] = (_Float16)__sinf(s * x0);
            dst[3 + 6 * j + 1] = (_Float16)__sinf(s * x1);
            dst[3 + 6 * j + 2] = (_Float16)__sinf(s * x2);
            dst[3 + 6 * j + 3] = (_Float16)__cosf(s * x0);
            dst[3 + 6 * j + 4] = (_Float16)__cosf(s * x1);
            dst[3 + 6 * j + 5] = (_Float16)__cosf(s * x2);
            s *= 2.0f;
        }
#pragma unroll
        for (int c = 27; c < 32; ++c) dst[c] = (_Float16)0.0f;   // pad 155..159 of b3 input
    }
    pipe_barrier();

    // ---- software-pipelined layers: stage L+1 async while computing L ----
    async_stage<SZ_128x128>(wb1, ws + OFF_W11, tid);
    mlp_layer<2, 8, true>(act + 128, act, wb0, lane, row16);    // x_emb(64) -> 128
    pipe_barrier();

    async_stage<SZ_128x128>(wb0, ws + OFF_W12, tid);
    mlp_layer<4, 8, true>(act, act, wb1, lane, row16);
    pipe_barrier();

    async_stage<SZ_128x128>(wb1, ws + OFF_W13, tid);
    mlp_layer<4, 8, true>(act, act, wb0, lane, row16);
    pipe_barrier();

    async_stage<SZ_128x128>(wb0, ws + OFF_W14, tid);
    mlp_layer<4, 8, true>(act, act, wb1, lane, row16);
    pipe_barrier();

    async_stage<SZ_192x128>(wb1, ws + OFF_W20, tid);
    mlp_layer<4, 8, true>(act, act, wb0, lane, row16);
    pipe_barrier();

    async_stage<SZ_128x128>(wb0, ws + OFF_W21, tid);
    mlp_layer<6, 8, true>(act, act, wb1, lane, row16);          // concat(h,x_emb) 192 -> 128
    pipe_barrier();

    async_stage<SZ_128x128>(wb1, ws + OFF_W22, tid);
    mlp_layer<4, 8, true>(act, act, wb0, lane, row16);
    pipe_barrier();

    async_stage<SZ_128x144>(wb0, ws + OFF_W23, tid);
    mlp_layer<4, 8, true>(act, act, wb1, lane, row16);
    pipe_barrier();

    // ---- b2_w3: 128 -> 129 (pad N to 144). Feat raw; sigma = relu(col 128). ----
    async_stage<SZ_160x64>(wb1, ws + OFF_W30, tid);
    {
        const int l15 = lane & 15;
        const int hi  = lane >> 4;
        const int row = row16 + l15;
        v16h a[4];
#pragma unroll
        for (int kt = 0; kt < 4; ++kt) a[kt] = ld_afrag(act, row, kt * 32, hi);
        v8f acc[9];
#pragma unroll
        for (int nt = 0; nt < 9; ++nt) acc[nt] = {};
#pragma unroll
        for (int kt = 0; kt < 4; ++kt) {
#pragma unroll
            for (int nt = 0; nt < 9; ++nt) {
                v16h b = ld_bfrag(wb0, kt * 9 + nt, lane);
                acc[nt] = __builtin_amdgcn_wmma_f32_16x16x32_f16(false, a[kt], false, b,
                                                                 (short)0, acc[nt], false, false);
            }
        }
#pragma unroll
        for (int nt = 0; nt < 8; ++nt)          // feat: raw (no relu)
#pragma unroll
            for (int i = 0; i < 8; i += 2)
                store_pair(act + (row16 + i + (hi << 3)) * ACT_W + nt * 16 + l15,
                           acc[nt][i], acc[nt][i + 1]);
        // sigma at global column 128 == (nt=8, N=0) -> lanes with l15==0
        if (l15 == 0) {
#pragma unroll
            for (int i = 0; i < 8; ++i) {
                const long long p = base + row16 + i + (hi << 3);
                if (p < nPts) sigmaOut[p] = relu1(acc[8][i]);
            }
        }
    }

    // ---- splice d_emb into cols 128..159 (wave-local rows, in-order LDS) ----
    {
        const int l15 = lane & 15;
        const int hi  = lane >> 4;
        const int row = row16 + l15;
        const _Float16* src = demb + row * 32 + (hi << 4);
        _Float16*       dst = act + row * ACT_W + 128 + (hi << 4);
        *(v8h*)dst       = *(const v8h*)src;
        *(v8h*)(dst + 8) = *(const v8h*)(src + 8);
    }
    pipe_barrier();

    // ---- block 3: concat(feat, d_emb) 160 -> 64 ----
    mlp_layer<5, 4, true>(act, act, wb1, lane, row16);

    // ---- final 64 -> 3 + sigmoid (tiny; VALU, uniform scalar weight loads) ----
    if (lane < 16) {
        const int r = row16 + lane;
        const long long p = base + r;
        float c0 = 0.0f, c1 = 0.0f, c2 = 0.0f;
#pragma unroll 8
        for (int k = 0; k < 64; ++k) {
            const float av = (float)act[r * ACT_W + k];
            c0 = fmaf(av, w31[k * 3 + 0], c0);
            c1 = fmaf(av, w31[k * 3 + 1], c1);
            c2 = fmaf(av, w31[k * 3 + 2], c2);
        }
        if (p < nPts) {
            colorOut[p * 3 + 0] = 1.0f / (1.0f + __expf(-c0));
            colorOut[p * 3 + 1] = 1.0f / (1.0f + __expf(-c1));
            colorOut[p * 3 + 2] = 1.0f / (1.0f + __expf(-c2));
        }
    }
}

extern "C" void kernel_launch(void* const* d_in, const int* in_sizes, int n_in,
                              void* d_out, int out_size, void* d_ws, size_t ws_size,
                              hipStream_t stream) {
    const float* pos  = (const float*)d_in[0];
    const float* dirs = (const float*)d_in[1];
    const float* w10  = (const float*)d_in[2];
    const float* w11  = (const float*)d_in[3];
    const float* w12  = (const float*)d_in[4];
    const float* w13  = (const float*)d_in[5];
    const float* w14  = (const float*)d_in[6];
    const float* w20  = (const float*)d_in[7];
    const float* w21  = (const float*)d_in[8];
    const float* w22  = (const float*)d_in[9];
    const float* w23  = (const float*)d_in[10];
    const float* w30  = (const float*)d_in[11];
    const float* w31  = (const float*)d_in[12];

    _Float16* ws = (_Float16*)d_ws;

    // one-time (per launch) weight swizzle into scratch; trivial cost
    preswizzle< 64, 128><<<SZ_64x128  / THREADS, THREADS, 0, stream>>>(w10, ws + OFF_W10,  63, 128);
    preswizzle<128, 128><<<SZ_128x128 / THREADS, THREADS, 0, stream>>>(w11, ws + OFF_W11, 128, 128);
    preswizzle<128, 128><<<SZ_128x128 / THREADS, THREADS, 0, stream>>>(w12, ws + OFF_W12, 128, 128);
    preswizzle<128, 128><<<SZ_128x128 / THREADS, THREADS, 0, stream>>>(w13, ws + OFF_W13, 128, 128);
    preswizzle<128, 128><<<SZ_128x128 / THREADS, THREADS, 0, stream>>>(w14, ws + OFF_W14, 128, 128);
    preswizzle<192, 128><<<SZ_192x128 / THREADS, THREADS, 0, stream>>>(w20, ws + OFF_W20, 191, 128);
    preswizzle<128, 128><<<SZ_128x128 / THREADS, THREADS, 0, stream>>>(w21, ws + OFF_W21, 128, 128);
    preswizzle<128, 128><<<SZ_128x128 / THREADS, THREADS, 0, stream>>>(w22, ws + OFF_W22, 128, 128);
    preswizzle<128, 144><<<SZ_128x144 / THREADS, THREADS, 0, stream>>>(w23, ws + OFF_W23, 128, 129);
    preswizzle<160,  64><<<SZ_160x64  / THREADS, THREADS, 0, stream>>>(w30, ws + OFF_W30, 155,  64);

    const int nPts = in_sizes[0] / 3;
    float* colorOut = (float*)d_out;               // (color[N,3], sigma[N]) flat
    float* sigmaOut = colorOut + (size_t)nPts * 3;

    const int blocks = (nPts + ROWS - 1) / ROWS;
    const size_t lds = (size_t)(ROWS * ACT_W + ROWS * 32 + 2 * WBUF_HALVES) * sizeof(_Float16);

    nerf_fused<<<dim3(blocks), dim3(THREADS), lds, stream>>>(
        pos, dirs, ws, w31, colorOut, sigmaOut, nPts);
}
